// GRU_27212912787836
// MI455X (gfx1250) — matrix operand order repaired and checked by reading, not compiled, not run on hardware
//
#include <hip/hip_runtime.h>

typedef _Float16 h16;
typedef unsigned int u32;
typedef __attribute__((ext_vector_type(8)))  _Float16 v8h;
typedef __attribute__((ext_vector_type(16))) _Float16 v16h;
typedef __attribute__((ext_vector_type(8)))  float    v8f;
typedef __attribute__((ext_vector_type(4)))  u32      u32x4;
typedef __attribute__((ext_vector_type(8)))  u32      u32x8;

constexpr int Bsz = 512, Tsz = 1024, H = 64;
constexpr int HP  = 72;   // padded LDS row stride (halves): 128B row + 16B pad
constexpr int OP  = 68;   // padded LDS row stride (floats) for the f32 output stage
constexpr int BT  = 16;   // batch rows per workgroup
constexpr int NW  = 4;    // waves per workgroup

__device__ __forceinline__ float fastrcp(float v) {
#if __has_builtin(__builtin_amdgcn_rcpf)
  return __builtin_amdgcn_rcpf(v);
#else
  return 1.0f / v;
#endif
}
__device__ __forceinline__ float fast_sigmoid(float v) {
  return fastrcp(1.0f + __expf(-v));              // v_exp_f32 + v_rcp_f32, branch-free
}
__device__ __forceinline__ float fast_tanh(float v) {
#if __has_builtin(__builtin_amdgcn_tanhf)
  return __builtin_amdgcn_tanhf(v);
#else
  return 1.0f - 2.0f * fastrcp(1.0f + __expf(2.0f * v));
#endif
}

// A fragment (16x32 f16): lane L<16 -> row L, K = {kb..kb+15}; lanes 16-31 shift K by 16.
__device__ __forceinline__ v16h load_a_frag(const h16* rowp, int kbase, int hi8) {
  const v8h lo = *(const v8h*)(rowp + kbase + 2*hi8);
  const v8h hh = *(const v8h*)(rowp + kbase + 8 + 2*hi8);
  return __builtin_shufflevector(lo, hh, 0,1,2,3,4,5,6,7,8,9,10,11,12,13,14,15);
}
// B fragment (32x16 f16) from the LDS f16 weight table: lane = column,
// elements = 16 contiguous K values at kpos -> two aligned b128 loads.
__device__ __forceinline__ v16h load_b_lds(const h16* rowp, int kpos) {
  const v8h lo = *(const v8h*)(rowp + kpos);
  const v8h hh = *(const v8h*)(rowp + kpos + 8);
  return __builtin_shufflevector(lo, hh, 0,1,2,3,4,5,6,7,8,9,10,11,12,13,14,15);
}
__device__ __forceinline__ v8f wmma_f16(v16h a, v16h b, v8f c) {
  return __builtin_amdgcn_wmma_f32_16x16x32_f16(false, a, false, b, (short)0, c, false, false);
}

__global__ __launch_bounds__(128) void gru2_kernel(
    const float* __restrict__ x,
    const float* __restrict__ Wih0, const float* __restrict__ Whh0,
    const float* __restrict__ bih0, const float* __restrict__ bhh0,
    const float* __restrict__ Wih1, const float* __restrict__ Whh1,
    const float* __restrict__ bih1, const float* __restrict__ bhh1,
    float* __restrict__ out, h16* __restrict__ h1ws)
{
  __shared__ h16   wtab[2][3*H][HP];  // f16 weight tables (layer0: [0]=W_hh0; layer1: [0]=W_ih1,[1]=W_hh1)
  __shared__ h16   hbuf[2][BT][HP];   // ping-pong hidden state (WMMA A operand)
  __shared__ h16   h1buf[2][BT][HP];  // double-buffered layer-1 input slice (filled by TDM)
  __shared__ float xbuf[BT][4];       // layer-0 x time slice
  __shared__ float obuf[2][BT][OP];   // ping-pong f32 output stage (drained by async stores)

  const int tid  = threadIdx.x;
  const int wv   = tid >> 5;
  const int lane = tid & 31;
  const int hi   = lane >> 4;
  const int hi8  = hi * 8;
  const int hi16 = hi * 16;
  const int b0   = blockIdx.x * BT;
  const int hc   = wv * 16 + (lane & 15);  // hidden column owned by this lane
  const int mrow = lane & 15;              // A-fragment row for this lane
  const int srow = tid >> 3;               // cooperative store: tile row
  const int schk = tid & 7;                // cooperative store: 16B chunk in row

  const unsigned hb0lds = (unsigned)(uintptr_t)&hbuf[0][0][0];
  const unsigned h1lds  = (unsigned)(uintptr_t)&h1buf[0][0][0];
  const unsigned oblds  = (unsigned)(uintptr_t)&obuf[0][0][0];

  //==================== layer 0 setup ====================
  for (int i = tid; i < 2 * BT * HP; i += 128) (&hbuf[0][0][0])[i] = (h16)0;
  for (int i = tid; i < 3*H*H; i += 128) {            // stage f16(W_hh0) into LDS
    wtab[0][i >> 6][i & 63] = (h16)Whh0[i];
  }
  __syncthreads();

  float wx[3][3], bsum0[2], bi0n, bh0n;
#pragma unroll
  for (int g = 0; g < 3; ++g) {
    const int gc = g * H + hc;
    wx[g][0] = Wih0[gc*3+0]; wx[g][1] = Wih0[gc*3+1]; wx[g][2] = Wih0[gc*3+2];
    if (g < 2) bsum0[g] = bih0[gc] + bhh0[gc];
  }
  bi0n = bih0[2*H+hc]; bh0n = bhh0[2*H+hc];

  v16h Bh0[3][2];                    // W_hh0 fragments, born from packed b128 LDS loads
#pragma unroll
  for (int g = 0; g < 3; ++g) {
    Bh0[g][0] = load_b_lds(&wtab[0][g*H + hc][0], 0  + hi16);
    Bh0[g][1] = load_b_lds(&wtab[0][g*H + hc][0], 32 + hi16);
  }

  float hreg[8];
#pragma unroll
  for (int i = 0; i < 8; ++i) hreg[i] = 0.f;

  const int xr = tid / 3, xf = tid - 3 * xr;          // x loader mapping (tid < 48)
  float xv = 0.f;
  if (tid < BT*3) xv = x[((b0 + xr)*Tsz + 0)*3 + xf]; // prefetch x(t=0)

  int p = 0;
  for (int t = 0; t < Tsz; ++t) {
    if (tid < BT*3) xbuf[xr][xf] = xv;                // publish prefetched x(t)
    // async-store previous step's hidden tile (reads hbuf[p]) -> h1 stream; 1-deep pipeline
    if (t > 0) {
      const unsigned sl = hb0lds + (unsigned)p*(BT*HP*2) + (unsigned)(srow*(HP*2) + schk*16);
      const unsigned so = ((((unsigned)(b0+srow))*Tsz + (unsigned)(t-1))*H + (unsigned)schk*8u) * 2u;
      asm volatile("global_store_async_from_lds_b128 %0, %1, %2" :: "v"(so), "v"(sl), "s"(h1ws) : "memory");
    }
    asm volatile("s_wait_asynccnt 0x1" ::: "memory"); // store(t-1) retired before hbuf[p] is rewritten
    __syncthreads();

    const h16* hrow = &hbuf[p][mrow][0];
    v16h a0 = load_a_frag(hrow, 0,  hi8);
    v16h a1 = load_a_frag(hrow, 32, hi8);
    v8f ar = {}, az = {}, an = {};
    ar = wmma_f16(a0, Bh0[0][0], ar); ar = wmma_f16(a1, Bh0[0][1], ar);
    az = wmma_f16(a0, Bh0[1][0], az); az = wmma_f16(a1, Bh0[1][1], az);
    an = wmma_f16(a0, Bh0[2][0], an); an = wmma_f16(a1, Bh0[2][1], an);
    if (tid < BT*3 && t + 1 < Tsz)                     // issue x(t+1) load early
      xv = x[((b0 + xr)*Tsz + (t+1))*3 + xf];
#pragma unroll
    for (int i = 0; i < 8; ++i) {
      const int m = i + hi8;
      const float x0 = xbuf[m][0], x1 = xbuf[m][1], x2 = xbuf[m][2];
      const float gr = fmaf(x0, wx[0][0], fmaf(x1, wx[0][1], x2*wx[0][2]));
      const float gz = fmaf(x0, wx[1][0], fmaf(x1, wx[1][1], x2*wx[1][2]));
      const float gn = fmaf(x0, wx[2][0], fmaf(x1, wx[2][1], x2*wx[2][2]));
      const float r  = fast_sigmoid(ar[i] + gr + bsum0[0]);
      const float z  = fast_sigmoid(az[i] + gz + bsum0[1]);
      const float n  = fast_tanh(gn + bi0n + r * (an[i] + bh0n));
      const float hn = (1.f - z)*n + z*hreg[i];
      hreg[i] = hn;
      hbuf[1-p][m][hc] = (h16)hn;
    }
    p ^= 1;
    __syncthreads();
  }
  { // flush h_{T-1}
    const unsigned sl = hb0lds + (unsigned)p*(BT*HP*2) + (unsigned)(srow*(HP*2) + schk*16);
    const unsigned so = ((((unsigned)(b0+srow))*Tsz + (unsigned)(Tsz-1))*H + (unsigned)schk*8u) * 2u;
    asm volatile("global_store_async_from_lds_b128 %0, %1, %2" :: "v"(so), "v"(sl), "s"(h1ws) : "memory");
    asm volatile("s_wait_asynccnt 0x0" ::: "memory");
  }
  __threadfence();
  __syncthreads();

  //==================== layer 1 setup ====================
  for (int i = tid; i < 2 * BT * HP; i += 128) (&hbuf[0][0][0])[i] = (h16)0;
  for (int i = tid; i < 3*H*H; i += 128) {            // stage f16(W_ih1), f16(W_hh1)
    wtab[0][i >> 6][i & 63] = (h16)Wih1[i];
    wtab[1][i >> 6][i & 63] = (h16)Whh1[i];
  }
#pragma unroll
  for (int i = 0; i < 8; ++i) hreg[i] = 0.f;
  __syncthreads();

  float bsum1[2];
  bsum1[0] = bih1[hc]   + bhh1[hc];
  bsum1[1] = bih1[H+hc] + bhh1[H+hc];
  const float bi1n = bih1[2*H+hc], bh1n = bhh1[2*H+hc];

  v16h Br[4], Bz[4], Bxn[2], Bhn[2];
#pragma unroll
  for (int kc = 0; kc < 2; ++kc) {
    Br[kc]   = load_b_lds(&wtab[0][hc][0],       kc*32 + hi16);
    Br[kc+2] = load_b_lds(&wtab[1][hc][0],       kc*32 + hi16);
    Bz[kc]   = load_b_lds(&wtab[0][H + hc][0],   kc*32 + hi16);
    Bz[kc+2] = load_b_lds(&wtab[1][H + hc][0],   kc*32 + hi16);
    Bxn[kc]  = load_b_lds(&wtab[0][2*H + hc][0], kc*32 + hi16);
    Bhn[kc]  = load_b_lds(&wtab[1][2*H + hc][0], kc*32 + hi16);
  }
  __syncthreads();

  // TDM descriptor group 1 (constant): f16 elements, 64x16 tile, row stride T*H,
  // LDS pad: 16B after every 128B so rows land on the HP=72 padded stride.
  const u32x8 g1v = { (1u<<16) | (1u<<20) | (4u<<22) | (3u<<25),
                      (64u << 16), (512u << 16), (64u << 16),
                      16u, (u32)(Tsz * H), 0u, 0u };
  const unsigned long long h1base =
      (unsigned long long)(uintptr_t)h1ws + ((unsigned long long)b0 * Tsz) * (H * 2ull);

  if (wv == 0) {                      // prologue: TDM prefetch slice t=0 into buffer 0
    const u32x4 g0v = { 1u, h1lds, (u32)h1base,
                        ((u32)(h1base >> 32) & 0x01FFFFFFu) | (2u << 30) };
    asm volatile("tensor_load_to_lds %0, %1" :: "s"(g0v), "s"(g1v) : "memory");
  }

  p = 0;
  for (int t = 0; t < Tsz; ++t) {
    if (wv == 0) {                    // TDM prefetch t+1, then wait for slice t (in-order)
      if (t + 1 < Tsz) {
        const unsigned long long ga = h1base + (unsigned long long)(t+1) * (H * 2ull);
        const u32x4 g0v = { 1u, h1lds + (u32)(((t+1)&1) * (BT*HP*2)), (u32)ga,
                            ((u32)(ga >> 32) & 0x01FFFFFFu) | (2u << 30) };
        asm volatile("tensor_load_to_lds %0, %1" :: "s"(g0v), "s"(g1v) : "memory");
        __builtin_amdgcn_s_wait_tensorcnt(1);
      } else {
        __builtin_amdgcn_s_wait_tensorcnt(0);
      }
    }
    if (t > 0) {                      // drain obuf[(t-1)&1]; 1-deep pipeline
#pragma unroll
      for (int q = 0; q < 2; ++q) {
        const unsigned c  = (unsigned)schk*2u + (unsigned)q;
        const unsigned sl = oblds + (unsigned)(((t-1)&1)*(BT*OP*4)) + (unsigned)srow*(OP*4u) + c*16u;
        const unsigned so = ((((unsigned)(b0+srow))*Tsz + (unsigned)(t-1))*H + c*4u) * 4u;
        asm volatile("global_store_async_from_lds_b128 %0, %1, %2" :: "v"(so), "v"(sl), "s"(out) : "memory");
      }
    }
    asm volatile("s_wait_asynccnt 0x1" ::: "memory");
    __syncthreads();

    const h16* hrow  = &hbuf[p][mrow][0];
    const h16* h1row = &h1buf[t & 1][mrow][0];
    v16h a0 = load_a_frag(h1row, 0,  hi8);
    v16h a1 = load_a_frag(h1row, 32, hi8);
    v16h a2 = load_a_frag(hrow,  0,  hi8);
    v16h a3 = load_a_frag(hrow,  32, hi8);
    v8f ar = {}, az = {}, axn = {}, ahn = {};
    ar  = wmma_f16(a0, Br[0], ar);   ar  = wmma_f16(a1, Br[1], ar);
    ar  = wmma_f16(a2, Br[2], ar);   ar  = wmma_f16(a3, Br[3], ar);
    az  = wmma_f16(a0, Bz[0], az);   az  = wmma_f16(a1, Bz[1], az);
    az  = wmma_f16(a2, Bz[2], az);   az  = wmma_f16(a3, Bz[3], az);
    axn = wmma_f16(a0, Bxn[0], axn); axn = wmma_f16(a1, Bxn[1], axn);
    ahn = wmma_f16(a2, Bhn[0], ahn); ahn = wmma_f16(a3, Bhn[1], ahn);
#pragma unroll
    for (int i = 0; i < 8; ++i) {
      const int m = i + hi8;
      const float r  = fast_sigmoid(ar[i] + bsum1[0]);
      const float z  = fast_sigmoid(az[i] + bsum1[1]);
      const float n  = fast_tanh(axn[i] + bi1n + r * (ahn[i] + bh1n));
      const float hn = (1.f - z)*n + z*hreg[i];
      hreg[i] = hn;
      hbuf[1-p][m][hc] = (h16)hn;
      obuf[t & 1][m][hc] = hn;
    }
    p ^= 1;
    __syncthreads();
  }
  { // flush t = T-1 output tile
#pragma unroll
    for (int q = 0; q < 2; ++q) {
      const unsigned c  = (unsigned)schk*2u + (unsigned)q;
      const unsigned sl = oblds + (unsigned)(((Tsz-1)&1)*(BT*OP*4)) + (unsigned)srow*(OP*4u) + c*16u;
      const unsigned so = ((((unsigned)(b0+srow))*Tsz + (unsigned)(Tsz-1))*H + c*4u) * 4u;
      asm volatile("global_store_async_from_lds_b128 %0, %1, %2" :: "v"(so), "v"(sl), "s"(out) : "memory");
    }
    asm volatile("s_wait_asynccnt 0x0" ::: "memory");
  }
  // final hidden = seq_out[:, T-1, :]
#pragma unroll
  for (int i = 0; i < 8; ++i) {
    const int m = i + hi8;
    out[(size_t)Bsz*Tsz*H + (size_t)(b0+m)*H + hc] = hreg[i];
  }
}

extern "C" void kernel_launch(void* const* d_in, const int* in_sizes, int n_in,
                              void* d_out, int out_size, void* d_ws, size_t ws_size,
                              hipStream_t stream) {
  (void)in_sizes; (void)n_in; (void)out_size; (void)ws_size;
  const float* x    = (const float*)d_in[0];
  const float* Wih0 = (const float*)d_in[1];
  const float* Whh0 = (const float*)d_in[2];
  const float* bih0 = (const float*)d_in[3];
  const float* bhh0 = (const float*)d_in[4];
  const float* Wih1 = (const float*)d_in[5];
  const float* Whh1 = (const float*)d_in[6];
  const float* bih1 = (const float*)d_in[7];
  const float* bhh1 = (const float*)d_in[8];
  float* out = (float*)d_out;
  h16* h1ws  = (h16*)d_ws;              // 512*1024*64 halves = 64 MB layer-1 input stream
  gru2_kernel<<<dim3(Bsz / BT), dim3(NW * 32), 0, stream>>>(
      x, Wih0, Whh0, bih0, bhh0, Wih1, Whh1, bih1, bhh1, out, h1ws);
}
